// PFNet7_14577119002739
// MI455X (gfx1250) — matrix operand order repaired and checked by reading, not compiled, not run on hardware
//
#include <hip/hip_runtime.h>
#include <hip/hip_bf16.h>
#include <math.h>

#define N_NODES 131072
#define KNN     16
#define IN_F    12
#define HID     32
#define ENC     256
#define SD      4
#define NID     6
#define NP4     4
#define P4_OFF  3

typedef __attribute__((ext_vector_type(16))) _Float16 v16h;
typedef __attribute__((ext_vector_type(8)))  float    v8f;

// ---- pre-packed B-fragment catalog: frag = base + ks*ntiles + nt ----
#define FR_WOUT 0    // 3 ksteps * 16 ntiles = 48
#define FR_W2_0 48   // 9 * 2 = 18
#define FR_W2_1 66   // 2
#define FR_W2_2 68   // 2
#define FR_W2_3 70   // 2
#define FR_W2_4 72   // 1
#define FR_W3_0 73   // 18
#define FR_W3_1 91   // 2
#define FR_W3_2 93   // 2
#define FR_W3_3 95   // 2
#define FR_W3_4 97   // 1
#define N_FRAGS 98   // 98 * 512 halves = 100,352 B (L2-resident)

// ---------------- WMMA fragment helpers (layouts per CDNA5 ISA 7.12.2) ----------------

// A fragment: 16x32 f16, row-major source in LDS with row stride `ld`.
__device__ __forceinline__ v16h load_A_lds(const _Float16* base, int ld, int k0) {
  const int lane = threadIdx.x & 31;
  const int m = lane & 15, hi = lane >> 4;
  v16h a;
#pragma unroll
  for (int e = 0; e < 16; ++e) {
    const int v = e >> 1;
    const int k = k0 + ((v & 4) ? 16 : 0) + hi * 8 + ((v & 3) << 1) + (e & 1);
    a[e] = base[m * ld + k];
  }
  return a;
}

// B fragment: one contiguous 32-byte per-lane load from the pre-packed buffer.
__device__ __forceinline__ v16h load_B_pack(const _Float16* Wp, int frag) {
  return *(const v16h*)(Wp + (size_t)frag * 512 + (threadIdx.x & 31) * 16);
}

__device__ __forceinline__ v8f wmma_f16(v16h a, v16h b, v8f c) {
  return __builtin_amdgcn_wmma_f32_16x16x32_f16(false, a, false, b, (short)0, c,
                                                false, false);
}

// Store a 16x16 D tile with bias + leaky-relu into an LDS activation [16][HID].
__device__ __forceinline__ void store_act_lrelu(v8f acc, const float* bias, int n0,
                                                _Float16* dst) {
  const int lane = threadIdx.x & 31;
  const int n = n0 + (lane & 15), hi = lane >> 4;
  const float bv = bias[n];
#pragma unroll
  for (int r = 0; r < 8; ++r) {
    const int m = r + hi * 8;
    float v = acc[r] + bv;
    v = v > 0.f ? v : 0.01f * v;
    dst[m * HID + n] = (_Float16)v;
  }
}

// ---------------- Kernel 0: pack weights into B-fragment layout (f16, padded) ----------------

struct PackParams {
  const float* W[11];
  int K[11];
  int Nout[11];
  int ntiles[11];
  int base[11];
};

__global__ __launch_bounds__(512) void k_pack(PackParams P, _Float16* __restrict__ Wp) {
  const int frag = blockIdx.x;
  int layer = 0;
#pragma unroll
  for (int l = 1; l < 11; ++l)
    if (frag >= P.base[l]) layer = l;
  const int local = frag - P.base[layer];
  const int nt = local % P.ntiles[layer];
  const int ks = local / P.ntiles[layer];
  const int lane = threadIdx.x >> 4, e = threadIdx.x & 15;
  const int k = ks * 32 + ((e & 8) ? 16 : 0) + ((lane & 16) ? 8 : 0) + (e & 7);   // k = k0 + 16*(e>=8) + 8*(lane>=16) + (e&7)
  const int n = nt * 16 + (lane & 15);
  const int K = P.K[layer], Nout = P.Nout[layer];
  const float v = (k < K && n < Nout) ? P.W[layer][k * Nout + n] : 0.f;
  Wp[(size_t)frag * 512 + threadIdx.x] = (_Float16)v;
}

// ---------------- Kernel 1: s = x@Ws+bs, h = x@Wh+bh ----------------

__global__ __launch_bounds__(256) void k_encode(const float* __restrict__ x,
                                                const float* __restrict__ Ws,
                                                const float* __restrict__ bs,
                                                const float* __restrict__ Wh,
                                                const float* __restrict__ bh,
                                                float* __restrict__ s,
                                                float* __restrict__ h) {
  const int i = blockIdx.x * 256 + threadIdx.x;
  float xv[IN_F];
#pragma unroll
  for (int c = 0; c < IN_F; ++c) xv[c] = x[i * IN_F + c];
#pragma unroll
  for (int o = 0; o < SD; ++o) {
    float acc = bs[o];
#pragma unroll
    for (int c = 0; c < IN_F; ++c) acc = fmaf(xv[c], Ws[c * SD + o], acc);
    s[i * SD + o] = acc;
  }
  for (int o = 0; o < HID; ++o) {
    float acc = bh[o];
#pragma unroll
    for (int c = 0; c < IN_F; ++c) acc = fmaf(xv[c], Wh[c * HID + o], acc);
    h[i * HID + o] = acc;
  }
}

// ---------------- Kernel 2: brute-force kNN with async double-buffered staging ----------------

#define CHC 512  // candidates per chunk; 2 buffers * 512 * float4 = 16 KB LDS

// Issue 2 async B128 copies per thread: one 8 KB chunk of candidate coords -> LDS.
__device__ __forceinline__ void knn_stage_async(const float4* src, float4* dst_lds) {
  const int tid = threadIdx.x;
#pragma unroll
  for (int t = 0; t < 2; ++t) {
    const uint32_t lds = (uint32_t)(uintptr_t)(dst_lds + tid + t * 256);
    const uint64_t g = (uint64_t)(uintptr_t)(src + tid + t * 256);
    asm volatile("global_load_async_to_lds_b128 %0, %1, off"
                 :: "v"(lds), "v"(g)
                 : "memory");
  }
}

__global__ __launch_bounds__(256) void k_knn(const float* __restrict__ s,
                                             int* __restrict__ knn) {
  __shared__ float4 cs[2][CHC];
  const int tid = threadIdx.x;
  const int q = blockIdx.x * 256 + tid;
  const float4 qs = ((const float4*)s)[q];

  float bd[KNN];
  int bi[KNN];
#pragma unroll
  for (int r = 0; r < KNN; ++r) { bd[r] = 3.0e38f; bi[r] = 0; }
  float worst = 3.0e38f;
  int wslot = 0;

  // prologue: stage chunk 0 into buffer 0 (2 async ops outstanding per wave)
  knn_stage_async((const float4*)s, &cs[0][0]);

  int buf = 0;
  for (int base = 0; base < N_NODES; base += CHC, buf ^= 1) {
    __syncthreads();  // all waves finished scanning cs[buf^1]; safe to overwrite it
    if (base + CHC < N_NODES) {
      // prefetch next chunk into the other buffer, then retire only the current
      // chunk's 2 ops (async loads complete in order -> asynccnt<=2 suffices)
      knn_stage_async((const float4*)s + base + CHC, &cs[buf ^ 1][0]);
      asm volatile("s_wait_asynccnt 0x2" ::: "memory");
    } else {
      asm volatile("s_wait_asynccnt 0x0" ::: "memory");
    }
    __syncthreads();  // publish every wave's stripe of the current chunk

    for (int j = 0; j < CHC; ++j) {
      const float4 c = cs[buf][j];
      const float dx = qs.x - c.x, dy = qs.y - c.y;
      const float dz = qs.z - c.z, dw = qs.w - c.w;
      const float d = dx * dx + dy * dy + dz * dz + dw * dw;
      if (d < worst) {  // running-threshold top-16: replace worst, rescan
        const int jj = base + j;
#pragma unroll
        for (int r = 0; r < KNN; ++r)
          if (r == wslot) { bd[r] = d; bi[r] = jj; }
        float wv = bd[0];
        int wsl = 0;
#pragma unroll
        for (int r = 1; r < KNN; ++r)
          if (bd[r] > wv) { wv = bd[r]; wsl = r; }
        worst = wv;
        wslot = wsl;
      }
    }
  }
#pragma unroll
  for (int r = 0; r < KNN; ++r) knn[q * KNN + r] = bi[r];
}

// ---------------- Kernel 3: fused gather + enc GEMM + both MLP heads ----------------

struct HeadBias {
  const float* b2[5];
  const float* b3[5];
};

__global__ __launch_bounds__(256) void k_fused(const float* __restrict__ x,
                                               const float* __restrict__ s,
                                               const float* __restrict__ h,
                                               const int* __restrict__ knn,
                                               const _Float16* __restrict__ Wp,
                                               const float* __restrict__ bout,
                                               HeadBias P,
                                               float* __restrict__ out) {
  __shared__ float    s_xq[16][IN_F];           // node features f32 (for p4 residual)
  __shared__ float    s_msg[16][KNN][HID];      // w * h_j messages  (32 KB)
  __shared__ _Float16 s_A[16][96];              // enc input [x|mean|max], K padded 76->96
  __shared__ _Float16 s_nnin[16][288];          // [x | x1 | ids | 0-pad], shared by nn2/nn3
  __shared__ _Float16 s_act[2][16][HID];        // MLP ping-pong activations

  const int tid = threadIdx.x;
  const int wave = tid >> 5;
  const int nb = blockIdx.x * 16;

  // ---- stage node features + zero the K-pad columns ----
  {
    const int i = tid >> 4, c = tid & 15;
    if (c < IN_F) {
      const float v = x[(size_t)(nb + i) * IN_F + c];
      s_xq[i][c] = v;
      s_A[i][c] = (_Float16)v;
      s_nnin[i][c] = (_Float16)v;
    }
    for (int cc = c; cc < 20; cc += 16) {
      s_A[i][76 + cc] = (_Float16)0.f;
      s_nnin[i][268 + cc] = (_Float16)0.f;
    }
  }

  // ---- GravNet gather: one (node, neighbor) pair per thread ----
  {
    const int i = tid >> 4, kk = tid & 15;
    const int node = nb + i;
    const int j = knn[node * KNN + kk];
    float d2 = 0.f;
#pragma unroll
    for (int c = 0; c < SD; ++c) {
      const float dd = s[node * SD + c] - s[j * SD + c];
      d2 += dd * dd;
    }
    const float w = __expf(-10.0f * d2);
#pragma unroll
    for (int c = 0; c < HID; ++c) s_msg[i][kk][c] = w * h[(size_t)j * HID + c];
  }
  __syncthreads();

  // ---- mean/max aggregation over K=16 neighbors ----
  {
    const int i = tid >> 4, c0 = tid & 15;
#pragma unroll
    for (int half = 0; half < 2; ++half) {
      const int c = c0 + half * 16;
      float mean = 0.f, mx = -3.0e38f;
#pragma unroll
      for (int kk = 0; kk < KNN; ++kk) {
        const float v = s_msg[i][kk][c];
        mean += v;
        mx = fmaxf(mx, v);
      }
      s_A[i][IN_F + c] = (_Float16)(mean * (1.0f / KNN));
      s_A[i][IN_F + HID + c] = (_Float16)mx;
    }
  }
  __syncthreads();

  // ---- enc = lrelu([x|mean|max] @ Wout + bout): 16x96 @ 96x256, 16 N-tiles / 8 waves ----
#pragma unroll
  for (int tt = 0; tt < 2; ++tt) {
    const int nt = wave + tt * 8;
    const int n0 = nt * 16;
    v8f acc = {};
#pragma unroll
    for (int ks = 0; ks < 3; ++ks) {
      const v16h a = load_A_lds(&s_A[0][0], 96, ks * 32);
      const v16h b = load_B_pack(Wp, FR_WOUT + ks * 16 + nt);
      acc = wmma_f16(a, b, acc);
    }
    const int lane = tid & 31, n = n0 + (lane & 15), hi = lane >> 4;
    const float bv = bout[n];
#pragma unroll
    for (int r = 0; r < 8; ++r) {
      const int m = r + hi * 8;
      float v = acc[r] + bv;
      v = v > 0.f ? v : 0.01f * v;
      s_nnin[m][IN_F + n] = (_Float16)v;   // x1 feeds both heads
    }
  }
  __syncthreads();

  // ================= nn2 head: [268 ->32 ->32 ->32 ->32 ->6] =================
  if (wave < 2) {
    v8f acc = {};
#pragma unroll
    for (int ks = 0; ks < 9; ++ks) {
      const v16h a = load_A_lds(&s_nnin[0][0], 288, ks * 32);
      const v16h b = load_B_pack(Wp, FR_W2_0 + ks * 2 + wave);
      acc = wmma_f16(a, b, acc);
    }
    store_act_lrelu(acc, P.b2[0], wave * 16, &s_act[0][0][0]);
  }
  __syncthreads();
#pragma unroll
  for (int L = 1; L < 4; ++L) {
    if (wave < 2) {
      const v16h a = load_A_lds(&s_act[(L - 1) & 1][0][0], HID, 0);
      const v16h b = load_B_pack(Wp, FR_W2_1 + (L - 1) * 2 + wave);
      v8f acc = {};
      acc = wmma_f16(a, b, acc);
      store_act_lrelu(acc, P.b2[L], wave * 16, &s_act[L & 1][0][0]);
    }
    __syncthreads();
  }
  if (wave == 0) {  // 32 -> 6 logits
    const v16h a = load_A_lds(&s_act[1][0][0], HID, 0);
    const v16h b = load_B_pack(Wp, FR_W2_4);
    v8f acc = {};
    acc = wmma_f16(a, b, acc);
    const int lane = tid & 31, n = lane & 15, hi = lane >> 4;
    if (n < NID) {
      const float bv = P.b2[4][n];
#pragma unroll
      for (int r = 0; r < 8; ++r) {
        const int m = r + hi * 8;
        const float v = acc[r] + bv;
        out[(size_t)(nb + m) * NID + n] = v;          // cand_ids
        s_nnin[m][IN_F + ENC + n] = (_Float16)v;      // feed nn3
      }
    }
  }
  __syncthreads();

  // ================= nn3 head: [274 ->32 ->32 ->32 ->32 ->4] =================
  if (wave < 2) {
    v8f acc = {};
#pragma unroll
    for (int ks = 0; ks < 9; ++ks) {
      const v16h a = load_A_lds(&s_nnin[0][0], 288, ks * 32);
      const v16h b = load_B_pack(Wp, FR_W3_0 + ks * 2 + wave);
      acc = wmma_f16(a, b, acc);
    }
    store_act_lrelu(acc, P.b3[0], wave * 16, &s_act[0][0][0]);
  }
  __syncthreads();
#pragma unroll
  for (int L = 1; L < 4; ++L) {
    if (wave < 2) {
      const v16h a = load_A_lds(&s_act[(L - 1) & 1][0][0], HID, 0);
      const v16h b = load_B_pack(Wp, FR_W3_1 + (L - 1) * 2 + wave);
      v8f acc = {};
      acc = wmma_f16(a, b, acc);
      store_act_lrelu(acc, P.b3[L], wave * 16, &s_act[L & 1][0][0]);
    }
    __syncthreads();
  }
  if (wave == 0) {  // 32 -> 4, plus p4 residual from x[:, 3:7]
    const v16h a = load_A_lds(&s_act[1][0][0], HID, 0);
    const v16h b = load_B_pack(Wp, FR_W3_4);
    v8f acc = {};
    acc = wmma_f16(a, b, acc);
    const int lane = tid & 31, n = lane & 15, hi = lane >> 4;
    if (n < NP4) {
      const float bv = P.b3[4][n];
#pragma unroll
      for (int r = 0; r < 8; ++r) {
        const int m = r + hi * 8;
        const float v = acc[r] + bv + s_xq[m][P4_OFF + n];
        out[(size_t)N_NODES * NID + (size_t)(nb + m) * NP4 + n] = v;  // cand_p4
      }
    }
  }
}

// ---------------- host launcher ----------------

extern "C" void kernel_launch(void* const* d_in, const int* in_sizes, int n_in,
                              void* d_out, int out_size, void* d_ws, size_t ws_size,
                              hipStream_t stream) {
  (void)in_sizes; (void)n_in; (void)out_size; (void)ws_size;
  const float* x    = (const float*)d_in[0];
  const float* Ws   = (const float*)d_in[1];
  const float* bs   = (const float*)d_in[2];
  const float* Wh   = (const float*)d_in[3];
  const float* bh   = (const float*)d_in[4];
  const float* Wout = (const float*)d_in[5];
  const float* bout = (const float*)d_in[6];

  HeadBias HB;
  for (int i = 0; i < 5; ++i) {
    HB.b2[i] = (const float*)d_in[12 + i];
    HB.b3[i] = (const float*)d_in[22 + i];
  }

  // Pack descriptors; layer order: Wout, w2_0..w2_4, w3_0..w3_4
  PackParams PP;
  const int bases[11] = {FR_WOUT, FR_W2_0, FR_W2_1, FR_W2_2, FR_W2_3, FR_W2_4,
                         FR_W3_0, FR_W3_1, FR_W3_2, FR_W3_3, FR_W3_4};
  const float* Wptr[11] = {Wout,
                           (const float*)d_in[7],  (const float*)d_in[8],
                           (const float*)d_in[9],  (const float*)d_in[10],
                           (const float*)d_in[11],
                           (const float*)d_in[17], (const float*)d_in[18],
                           (const float*)d_in[19], (const float*)d_in[20],
                           (const float*)d_in[21]};
  const int Kv[11] = {76, 268, 32, 32, 32, 32, 274, 32, 32, 32, 32};
  const int Nv[11] = {256, 32, 32, 32, 6, 32, 32, 32, 32, 4, 4};
  // NOTE: Nv must match layer order exactly: Wout=256, w2 layers = 32,32,32,32,6,
  // w3 layers = 32,32,32,32,4.
  const int Nv_fixed[11] = {256, 32, 32, 32, 32, 6, 32, 32, 32, 32, 4};
  const int NT[11] = {16, 2, 2, 2, 2, 1, 2, 2, 2, 2, 1};
  (void)Nv;
  for (int l = 0; l < 11; ++l) {
    PP.W[l] = Wptr[l];
    PP.K[l] = Kv[l];
    PP.Nout[l] = Nv_fixed[l];
    PP.ntiles[l] = NT[l];
    PP.base[l] = bases[l];
  }

  float* ws  = (float*)d_ws;
  float* s   = ws;                                         // N*4  f32
  float* h   = ws + (size_t)N_NODES * SD;                  // N*32 f32
  int*   knn = (int*)(ws + (size_t)N_NODES * (SD + HID));  // N*16 i32
  _Float16* Wp = (_Float16*)(ws + (size_t)N_NODES * (SD + HID + KNN));  // packed B frags
  float* out = (float*)d_out;                              // [N*6 ids | N*4 p4]

  k_pack<<<N_FRAGS, 512, 0, stream>>>(PP, Wp);
  k_encode<<<N_NODES / 256, 256, 0, stream>>>(x, Ws, bs, Wh, bh, s, h);
  k_knn<<<N_NODES / 256, 256, 0, stream>>>(s, knn);
  k_fused<<<N_NODES / 16, 256, 0, stream>>>(x, s, h, knn, Wp, bout, HB, out);
}
